// CrossAttention_24008867184861
// MI455X (gfx1250) — matrix operand order repaired and checked by reading, not compiled, not run on hardware
//
#include <hip/hip_runtime.h>

// Flash-attention: B=2, Sq=Sk=2048, H=16 (GQA Hkv=4), D=64, fp32 in/out,
// causal + key-padding mask. f16 WMMA 16x16x32, f32 accumulation.
// Block = 256 threads = 8 wave32; wave owns 16 query rows (block: 128 rows);
// 64 keys/iteration. Softmax in log2 domain; row-sum via WMMA with ones-B.

typedef __attribute__((ext_vector_type(16))) _Float16 v16h;
typedef __attribute__((ext_vector_type(8)))  _Float16 v8h;
typedef __attribute__((ext_vector_type(4)))  _Float16 v4h;
typedef __attribute__((ext_vector_type(8)))  float    v8f;
typedef __attribute__((ext_vector_type(4)))  float    v4f;

constexpr float LOG2E   = 1.44269504088896340736f;
constexpr float PADNEG  = -10000.0f * LOG2E;   // padding addend (log2 domain)
constexpr float CAUSNEG = -30000.0f * LOG2E;   // causal mask value (log2 domain)

constexpr int Bc = 2, Sq = 2048, Sk = 2048, H = 16, Hkv = 4, D = 64;
constexpr int QTILE = 128;     // query rows per block
constexpr int KTILE = 64;      // keys per iteration
constexpr int NWAVE = 8;
constexpr int NTHREADS = 256;

// hardware 2^x (v_exp_f32 is natively base-2)
__device__ __forceinline__ float fast_exp2(float x) {
  return __builtin_amdgcn_exp2f(x);
}

// lane-xor gather within 16-lane groups via ds_swizzle (no address VALU)
template <int MASK>
__device__ __forceinline__ float swz_xor(float v) {
  return __int_as_float(
      __builtin_amdgcn_ds_swizzle(__float_as_int(v), (MASK << 10) | 31));
}

// ---- one 64-key attention step for one wave (16 query rows) ----
template <bool CAUSAL>
__device__ __forceinline__ void attn_step(
    int k0, int rowMin, int col, int hi,
    const v16h (&qa)[2], const v16h& ones,
    v8f (&o)[4], float (&mrow)[8], float (&lrow)[8],
    const _Float16* __restrict__ sK, const _Float16* __restrict__ sVt,
    _Float16* __restrict__ pw, const float* __restrict__ sPad) {
  // S = Q K^T : four 16x16 f32 tiles (keys t*16 .. t*16+15), log2-scaled
  v8f s[4];
#pragma unroll
  for (int t = 0; t < 4; ++t) {
    v8f acc = {};
#pragma unroll
    for (int c = 0; c < 2; ++c) {   // head-dim chunks of 32
      v16h kb = *(const v16h*)(&sK[(t * 16 + col) * D + c * 32 + hi * 16]);
      acc = __builtin_amdgcn_wmma_f32_16x16x32_f16(false, qa[c], false, kb,
                                                   (short)0, acc, false, false);
    }
    s[t] = acc;
  }

  // key-padding addend for this lane's column in each tile
  float padv[4];
#pragma unroll
  for (int t = 0; t < 4; ++t) padv[t] = sPad[t * 16 + col];

  // online softmax: row max (reduce across the 16 columns via ds_swizzle)
  float mnew[8];
#pragma unroll
  for (int r = 0; r < 8; ++r) {
    const int row = rowMin + r + hi * 8;   // C layout: VGPR r -> row r / r+8
    float mx = mrow[r];
#pragma unroll
    for (int t = 0; t < 4; ++t) {
      float v = s[t][r] + padv[t];
      if (CAUSAL) {
        if (k0 + t * 16 + col > row) v = CAUSNEG;
      }
      s[t][r] = v;
      mx = fmaxf(mx, v);
    }
    mx = fmaxf(mx, swz_xor<1>(mx));
    mx = fmaxf(mx, swz_xor<2>(mx));
    mx = fmaxf(mx, swz_xor<4>(mx));
    mx = fmaxf(mx, swz_xor<8>(mx));
    mnew[r] = mx;
  }

  // exp2, rescale running output
  float alpha[8];
#pragma unroll
  for (int r = 0; r < 8; ++r) {
    alpha[r] = fast_exp2(mrow[r] - mnew[r]);
    mrow[r] = mnew[r];
#pragma unroll
    for (int t = 0; t < 4; ++t) s[t][r] = fast_exp2(s[t][r] - mnew[r]);
#pragma unroll
    for (int t = 0; t < 4; ++t) o[t][r] *= alpha[r];
  }

  // C-layout -> A-layout conversion of P via wave-private LDS
  // (same-wave DS ops are in-order per the ISA; no barrier needed)
#pragma unroll
  for (int t = 0; t < 4; ++t)
#pragma unroll
    for (int r = 0; r < 8; ++r)
      pw[(r + hi * 8) * KTILE + t * 16 + col] = (_Float16)s[t][r];

  v16h pa[2];
#pragma unroll
  for (int c = 0; c < 2; ++c) {     // key chunks of 32
    const _Float16* p0 = &pw[col * KTILE + c * 32 + hi * 16];
    v8h a0 = *(const v8h*)(p0);
    v8h a1 = *(const v8h*)(p0 + 8);
#pragma unroll
    for (int j = 0; j < 8; ++j) { pa[c][j] = a0[j]; pa[c][j + 8] = a1[j]; }
  }

  // row sums of P via WMMA against an all-ones B matrix (replaces shuffles)
  v8f rsum = {};
#pragma unroll
  for (int c = 0; c < 2; ++c)
    rsum = __builtin_amdgcn_wmma_f32_16x16x32_f16(false, pa[c], false, ones,
                                                  (short)0, rsum, false, false);
#pragma unroll
  for (int r = 0; r < 8; ++r) lrow[r] = lrow[r] * alpha[r] + rsum[r];

  // O += P * V : four output-column tiles x two key chunks
#pragma unroll
  for (int t = 0; t < 4; ++t) {
#pragma unroll
    for (int c = 0; c < 2; ++c) {
      v16h vb = *(const v16h*)(&sVt[(t * 16 + col) * KTILE + c * 32 + hi * 16]);
      o[t] = __builtin_amdgcn_wmma_f32_16x16x32_f16(false, pa[c], false, vb,
                                                    (short)0, o[t], false, false);
    }
  }
}

// ---- stage a 64-key K/V tile (fp32 -> f16; V transposed) ----
__device__ __forceinline__ void stage_kv(
    int k0, int tid, int hk, const float* __restrict__ kvb,
    const unsigned char* __restrict__ mask, size_t maskBase,
    _Float16* __restrict__ sK, _Float16* __restrict__ sVt,
    float* __restrict__ sPad) {
#pragma unroll
  for (int i = 0; i < 4; ++i) {
    int idx4 = tid + i * NTHREADS;    // 1024 float4 = 64x64 floats
    int base = idx4 * 4;
    int kk = base >> 6, d = base & 63;
    const float* rowp = kvb + (size_t)(k0 + kk) * (2 * Hkv * D) + hk * D + d;
    v4f kq = *(const v4f*)(rowp);
    v4f vq = *(const v4f*)(rowp + Hkv * D);
    v4h kh;
#pragma unroll
    for (int j = 0; j < 4; ++j) kh[j] = (_Float16)kq[j];
    *(v4h*)(&sK[kk * D + d]) = kh;
#pragma unroll
    for (int j = 0; j < 4; ++j) sVt[(d + j) * KTILE + kk] = (_Float16)vq[j];
  }
  if (tid < KTILE)
    sPad[tid] = mask[maskBase + k0 + tid] ? 0.0f : PADNEG;
}

__global__ __launch_bounds__(NTHREADS) void
fa_fwd_kernel(const float* __restrict__ q,
              const float* __restrict__ kv,
              const unsigned char* __restrict__ mask,  // jax bool -> 1 byte
              float* __restrict__ out) {
  __shared__ __align__(32) _Float16 sQ[QTILE * D];          // [128][64] pre-scaled
  __shared__ __align__(32) _Float16 sK[KTILE * D];          // [64 keys][64 dims]
  __shared__ __align__(32) _Float16 sVt[D * KTILE];         // [64 dims][64 keys]
  __shared__ __align__(32) _Float16 sP[NWAVE][16 * KTILE];  // per-wave P staging
  __shared__ float sPad[KTILE];

  const int tid  = threadIdx.x;
  const int wave = tid >> 5;
  const int lane = tid & 31;
  const int col  = lane & 15;
  const int hi   = lane >> 4;

  const int bh = blockIdx.y;     // b*H + h
  const int b  = bh / H;
  const int h  = bh % H;
  const int hk = h / (H / Hkv);
  const int q0 = blockIdx.x * QTILE;
  const float scale = 0.125f * LOG2E;   // 1/sqrt(64), log2 domain

  // ---- stage Q tile (128x64) as f16, pre-scaled ----
  {
    const float* qbase = q + (((size_t)b * Sq + q0) * H + h) * D;
#pragma unroll
    for (int i = 0; i < 8; ++i) {
      int idx4 = tid + i * NTHREADS;    // 2048 float4
      int base = idx4 * 4;
      int r = base >> 6, d = base & 63;
      v4f qv = *(const v4f*)(qbase + (size_t)r * H * D + d);
      v4h qh;
#pragma unroll
      for (int j = 0; j < 4; ++j) qh[j] = (_Float16)(qv[j] * scale);
      *(v4h*)(&sQ[r * D + d]) = qh;
    }
  }
  __syncthreads();

  // per-wave Q A-fragments: lane m=lane%16 holds K={0-15} / {16-31}
  v16h qa[2];
  {
    const int row = wave * 16 + col;
#pragma unroll
    for (int c = 0; c < 2; ++c) {
      const _Float16* p0 = &sQ[row * D + c * 32 + hi * 16];
      v8h a0 = *(const v8h*)(p0);
      v8h a1 = *(const v8h*)(p0 + 8);
#pragma unroll
      for (int j = 0; j < 8; ++j) { qa[c][j] = a0[j]; qa[c][j + 8] = a1[j]; }
    }
  }
  v16h ones;
#pragma unroll
  for (int j = 0; j < 16; ++j) ones[j] = (_Float16)1.0f;

  v8f o[4];
#pragma unroll
  for (int t = 0; t < 4; ++t) o[t] = {};
  float mrow[8], lrow[8];
#pragma unroll
  for (int r = 0; r < 8; ++r) { mrow[r] = -3.0e38f; lrow[r] = 0.0f; }

  const int rowMin = q0 + wave * 16;
  const int rowMax = rowMin + 15;
  const int nIter  = 2 * blockIdx.x + 2;   // causal bound: keys <= q0+127
  const float* kvb = kv + (size_t)b * Sk * 2 * Hkv * D;
  const size_t maskBase = (size_t)b * Sk;
  _Float16* pw = sP[wave];

  // ---- steady state: k0+63 < rowMin for all waves -> no causal masking ----
  for (int it = 0; it < nIter - 2; ++it) {
    const int k0 = it * KTILE;
    __syncthreads();                      // previous tile's consumers done
    stage_kv(k0, tid, hk, kvb, mask, maskBase, sK, sVt, sPad);
    __syncthreads();
    if (tid < 2 * KTILE) {                // prefetch next K/V tile rows
      const float* pf = kvb + (size_t)(k0 + KTILE + (tid >> 1)) * (2 * Hkv * D)
                            + (size_t)(tid & 1) * (Hkv * D) + hk * D;
      __builtin_prefetch(pf, 0, 3);       // -> global_prefetch_b8
    }
    attn_step<false>(k0, rowMin, col, hi, qa, ones, o, mrow, lrow,
                     sK, sVt, pw, sPad);
  }

  // ---- last two tiles: diagonal region, causal masking + full-skip ----
#pragma unroll
  for (int e = 2; e >= 1; --e) {
    const int it = nIter - e;
    const int k0 = it * KTILE;
    __syncthreads();
    stage_kv(k0, tid, hk, kvb, mask, maskBase, sK, sVt, sPad);
    __syncthreads();
    if (e == 2 && tid < 2 * KTILE) {      // prefetch final tile
      const float* pf = kvb + (size_t)(k0 + KTILE + (tid >> 1)) * (2 * Hkv * D)
                            + (size_t)(tid & 1) * (Hkv * D) + hk * D;
      __builtin_prefetch(pf, 0, 3);
    }
    if (k0 <= rowMax)                     // wave-uniform: skip fully-masked tile
      attn_step<true>(k0, rowMin, col, hi, qa, ones, o, mrow, lrow,
                      sK, sVt, pw, sPad);
  }

  // ---- normalize and write out (fp32, bthd layout) ----
#pragma unroll
  for (int r = 0; r < 8; ++r) {
    const int row = rowMin + r + hi * 8;
    const float inv = 1.0f / lrow[r];
#pragma unroll
    for (int t = 0; t < 4; ++t) {
      const int d = t * 16 + col;
      out[(((size_t)b * Sq + row) * H + h) * D + d] = o[t][r] * inv;
    }
  }
}

extern "C" void kernel_launch(void* const* d_in, const int* in_sizes, int n_in,
                              void* d_out, int out_size, void* d_ws, size_t ws_size,
                              hipStream_t stream) {
  const float* q  = (const float*)d_in[0];
  const float* kv = (const float*)d_in[1];
  const unsigned char* mask = (const unsigned char*)d_in[2];
  float* out = (float*)d_out;

  dim3 grid(Sq / QTILE, Bc * H);   // (16, 32)
  dim3 block(NTHREADS);
  fa_fwd_kernel<<<grid, block, 0, stream>>>(q, kv, mask, out);
}
